// DecoderBlock_5334349382405
// MI455X (gfx1250) — matrix operand order repaired and checked by reading, not compiled, not run on hardware
//
#include <hip/hip_runtime.h>

// ---------------------------------------------------------------------------
// Transformer decoder block for MI455X (gfx1250, wave32, WMMA).
// All matmuls: v_wmma_f32_16x16x32_bf16 (bf16 in, f32 accumulate).
// Dense GEMMs: 4-wave 128x64 block tiles, A/B panels staged into LDS with
// global_load_async_to_lds_b128 (double-buffered, s_wait_asynccnt), K-loop
// fully unrolled via template.
// Attention is flash-style: per-wave 16 query rows, scores kept in LDS.
// ---------------------------------------------------------------------------

#define B_   128
#define S_   256
#define D_   256
#define H_   6
#define HS_  42          // D/H
#define HSP_ 64          // HS padded to a multiple of 32 for WMMA K-dim
#define DFF_ 1024
#define BS_  (B_ * S_)
#define HCAT_ (H_ * HS_) // 252

typedef __bf16 bf16;
typedef __attribute__((ext_vector_type(16))) __bf16 v16bf;
typedef __attribute__((ext_vector_type(8)))  __bf16 v8bf;
typedef __attribute__((ext_vector_type(8)))  float  v8f;

// ---------------------------------------------------------------------------
// WMMA helpers
// ---------------------------------------------------------------------------
__device__ __forceinline__ v8f wmma_bf16(v16bf a, v16bf b, v8f c) {
  // D = A(16x32) * B(32x16) + C(16x16 f32)
  return __builtin_amdgcn_wmma_f32_16x16x32_bf16(
      /*neg_a=*/false, a, /*neg_b=*/false, b,
      /*c_mod=*/(short)0, c, /*reuse_a=*/false, /*reuse_b=*/false);
}

// A-matrix 16x32 bf16 fragment (ISA 7.12.2): lane holds row (lane&15);
// elements 0..7 = K[base..base+7], 8..15 = K[16+base..16+base+7], base = (lane<16)?0:8.
// p points at element (row0, k0) of a row-major matrix with leading dim ld.
__device__ __forceinline__ v16bf load_a_frag(const bf16* p, int ld, int lane) {
  const int m    = lane & 15;
  const int base = (lane & 16) ? 8 : 0;
  const bf16* rp = p + (size_t)m * ld;
  v8bf lo = *(const v8bf*)(rp + base);
  v8bf hi = *(const v8bf*)(rp + 16 + base);
  v16bf a;
#pragma unroll
  for (int e = 0; e < 8; ++e) { a[e] = lo[e]; a[e + 8] = hi[e]; }
  return a;
}

// B-matrix 32x16 bf16 fragment from TRANSPOSED storage ([N][K] row-major, ld = K
// stride): lane holds column n = lane&15; elements 0..7 = K[base..base+7], 8..15 = K[16+base..16+base+7], base = (lane<16)?0:8.
// Transposed packing makes each lane's K-elements contiguous in 8-runs (2x b128 loads).
__device__ __forceinline__ v16bf load_b_frag(const bf16* p, int ld, int lane) {
  const int n  = lane & 15;
  const int kb = (lane & 16) ? 8 : 0;
  const bf16* rp = p + (size_t)n * ld + kb;
  v8bf lo = *(const v8bf*)(rp);
  v8bf hi = *(const v8bf*)(rp + 16);
  v16bf b;
#pragma unroll
  for (int e = 0; e < 8; ++e) { b[e] = lo[e]; b[e + 8] = hi[e]; }
  return b;
}

// ---------------------------------------------------------------------------
// CDNA5 async global->LDS DMA (ASYNCcnt-tracked), via inline asm so it is
// portable across ROCm/upstream builtin arities. Each call moves 16 bytes.
// ---------------------------------------------------------------------------
__device__ __forceinline__ uint32_t lds_addr_of(const void* p) {
  return (uint32_t)(size_t)(__attribute__((address_space(3))) const void*)p;
}

__device__ __forceinline__ void async_copy_b128(uint32_t lds_off, const void* gptr) {
  asm volatile("global_load_async_to_lds_b128 %0, %1, off"
               :: "v"(lds_off), "v"(gptr) : "memory");
}

template <int N>
__device__ __forceinline__ void wait_asynccnt() {
  asm volatile("s_wait_asynccnt %0" :: "n"(N) : "memory");
}

// ---------------------------------------------------------------------------
// Prep kernels: f32 -> bf16 conversion, transposed + zero-padded weight packs
// ---------------------------------------------------------------------------
__global__ void pack_x_kernel(const float* __restrict__ x, bf16* __restrict__ xb, int n) {
  int i = blockIdx.x * blockDim.x + threadIdx.x;
  if (i < n) xb[i] = (bf16)x[i];
}

// Wq/Wk/Wv [H, D, HS] -> WT [H, HSP, D] (transposed per head, HS padded to HSP w/ zeros)
__global__ void pack_qkvw_kernel(const float* __restrict__ Wq, const float* __restrict__ Wk,
                                 const float* __restrict__ Wv, bf16* __restrict__ WqT,
                                 bf16* __restrict__ WkT, bf16* __restrict__ WvT) {
  int i = blockIdx.x * blockDim.x + threadIdx.x;
  if (i >= H_ * HSP_ * D_) return;
  int k = i % D_;
  int n = (i / D_) % HSP_;
  int h = i / (D_ * HSP_);
  float vq = 0.f, vk = 0.f, vv = 0.f;
  if (n < HS_) {
    size_t s = (size_t)h * D_ * HS_ + (size_t)k * HS_ + n;
    vq = Wq[s]; vk = Wk[s]; vv = Wv[s];
  }
  WqT[i] = (bf16)vq; WkT[i] = (bf16)vk; WvT[i] = (bf16)vv;
}

// Wproj [252, 256] -> WT [256][256] transposed, zero-padded K rows 252..255
__global__ void pack_proj_kernel(const float* __restrict__ W, bf16* __restrict__ WT) {
  int i = blockIdx.x * blockDim.x + threadIdx.x;
  if (i >= D_ * D_) return;
  int k = i % D_;
  int n = i / D_;
  WT[i] = (bf16)((k < HCAT_) ? W[(size_t)k * D_ + n] : 0.f);
}

// Generic transpose pack: W [K, N] f32 -> WT [N, K] bf16
__global__ void pack_t_kernel(const float* __restrict__ W, bf16* __restrict__ WT, int K, int N) {
  int i = blockIdx.x * blockDim.x + threadIdx.x;
  if (i >= K * N) return;
  int k = i % K;
  int n = i / K;
  WT[i] = (bf16)W[(size_t)k * N + n];
}

// head_cat has 252 valid cols; zero the 4 pad cols once
__global__ void zero_hc_tail_kernel(bf16* __restrict__ hc) {
  int i = blockIdx.x * blockDim.x + threadIdx.x;
  if (i >= BS_ * 4) return;
  int row = i >> 2, c = HCAT_ + (i & 3);
  hc[(size_t)row * D_ + c] = (bf16)0.f;
}

// ---------------------------------------------------------------------------
// QKV projection: per (b,h): q/k = x(16xD) @ W(Dx16-tile); v stored transposed.
// grid = (S/16, HSP/16, B*H), block = 1 wave
// ---------------------------------------------------------------------------
__global__ __launch_bounds__(32) void qkv_kernel(
    const bf16* __restrict__ xb, const bf16* __restrict__ WqT,
    const bf16* __restrict__ WkT, const bf16* __restrict__ WvT,
    bf16* __restrict__ q, bf16* __restrict__ k, bf16* __restrict__ vT) {
  const int lane  = threadIdx.x;
  const int stile = blockIdx.x;           // row tile over S
  const int ntile = blockIdx.y;           // col tile over HSP
  const int bh    = blockIdx.z;
  const int b     = bh / H_;
  const int h     = bh % H_;

  const bf16* Ar = xb + ((size_t)b * S_ + (size_t)stile * 16) * D_;
  const bf16* Bq = WqT + ((size_t)h * HSP_ + (size_t)ntile * 16) * D_;
  const bf16* Bk = WkT + ((size_t)h * HSP_ + (size_t)ntile * 16) * D_;
  const bf16* Bv = WvT + ((size_t)h * HSP_ + (size_t)ntile * 16) * D_;

  v8f cq = {}, ck = {}, cv = {};
#pragma unroll
  for (int k0 = 0; k0 < D_; k0 += 32) {
    __builtin_prefetch(Ar + k0 + 64, 0, 3);     // global_prefetch_b8
    v16bf a = load_a_frag(Ar + k0, D_, lane);
    cq = wmma_bf16(a, load_b_frag(Bq + k0, D_, lane), cq);
    ck = wmma_bf16(a, load_b_frag(Bk + k0, D_, lane), ck);
    cv = wmma_bf16(a, load_b_frag(Bv + k0, D_, lane), cv);
  }

  const int n  = lane & 15;
  const int mb = (lane & 16) ? 8 : 0;
  // q,k: [bh][S][HSP] row-major; vT: [bh][HSP][S] (transposed for B-frag reads)
  const size_t qkbase = ((size_t)bh * S_ + (size_t)stile * 16) * HSP_ + ntile * 16 + n;
  const size_t vbase  = ((size_t)bh * HSP_ + (size_t)ntile * 16 + n) * S_ + (size_t)stile * 16;
#pragma unroll
  for (int r = 0; r < 8; ++r) {
    const int m = mb + r;
    q[qkbase + (size_t)m * HSP_] = (bf16)cq[r];
    k[qkbase + (size_t)m * HSP_] = (bf16)ck[r];
    vT[vbase + m]                = (bf16)cv[r];
  }
}

// ---------------------------------------------------------------------------
// Causal attention, flash-style. One wave owns 16 query rows; the 16xS score
// strip lives in LDS (16 KB), softmax in LDS, P@V via WMMA. Writes head_cat
// (bf16, [BS][256], head h at cols h*42..h*42+41).
// grid = (S/16, B*H), block = 1 wave
// ---------------------------------------------------------------------------
__global__ __launch_bounds__(32) void attn_kernel(
    const bf16* __restrict__ q, const bf16* __restrict__ k,
    const bf16* __restrict__ vT, bf16* __restrict__ hc) {
  __shared__ __align__(16) float sl[16][S_];    // 16 KB  raw scores / exp values
  __shared__ __align__(16) bf16  pl[16][S_];    //  8 KB  normalized probs (A-frag source)

  const int lane  = threadIdx.x;
  const int itile = blockIdx.x;
  const int bh    = blockIdx.y;
  const int b     = bh / H_;
  const int h     = bh % H_;
  const int n     = lane & 15;
  const int mb    = (lane & 16) ? 8 : 0;
  const float scale = 0.15430334996209191f;   // 1/sqrt(42)

  const bf16* qp = q + ((size_t)bh * S_ + (size_t)itile * 16) * HSP_;
  const v16bf a0 = load_a_frag(qp, HSP_, lane);        // K = 0..31 of HSP
  const v16bf a1 = load_a_frag(qp + 32, HSP_, lane);   // K = 32..63 (zero-padded)

  // ---- scores: S_tile = Q(16xHSP) @ K^T, causal-masked, into LDS ----
  for (int j = 0; j <= itile; ++j) {
    const bf16* kp = k + ((size_t)bh * S_ + (size_t)j * 16) * HSP_; // [t][HSP] == B^T[n][k]
    v8f c = {};
    c = wmma_bf16(a0, load_b_frag(kp, HSP_, lane), c);
    c = wmma_bf16(a1, load_b_frag(kp + 32, HSP_, lane), c);
#pragma unroll
    for (int r = 0; r < 8; ++r) {
      const int m = mb + r;
      float sc = c[r] * scale;
      if (j == itile && n > m) sc = -1e30f;   // causal mask inside diagonal tile
      sl[m][j * 16 + n] = sc;
    }
  }
  // zero prob strip (covers the odd-tile tail of the last 32-wide A chunk)
  {
    bf16* pf = &pl[0][0];
    for (int i = lane; i < 16 * S_; i += 32) pf[i] = (bf16)0.f;
  }
  __syncthreads();

  // ---- softmax over valid columns, wave-cooperative per row ----
  const int ncols = (itile + 1) * 16;
  for (int r = 0; r < 16; ++r) {
    float mx = -1e30f;
    for (int c = lane; c < ncols; c += 32) mx = fmaxf(mx, sl[r][c]);
#pragma unroll
    for (int o = 16; o; o >>= 1) mx = fmaxf(mx, __shfl_xor(mx, o, 32));
    float sum = 0.f;
    for (int c = lane; c < ncols; c += 32) {
      float e = __expf(sl[r][c] - mx);
      sl[r][c] = e;
      sum += e;
    }
#pragma unroll
    for (int o = 16; o; o >>= 1) sum += __shfl_xor(sum, o, 32);
    const float inv = 1.f / sum;
    for (int c = lane; c < ncols; c += 32) pl[r][c] = (bf16)(sl[r][c] * inv);
  }
  __syncthreads();

  // ---- O = P(16 x ncols) @ V(ncols x HSP) via WMMA, K-chunks of 32 ----
  v8f o0 = {}, o1 = {}, o2 = {};
  const bf16* vp = vT + (size_t)bh * HSP_ * S_;   // [HSP][S] transposed
  const int nch = (ncols + 31) / 32;
  for (int ch = 0; ch < nch; ++ch) {
    const v16bf ap = load_a_frag(&pl[0][0] + ch * 32, S_, lane);  // LDS reads
    o0 = wmma_bf16(ap, load_b_frag(vp + (size_t)0  * S_ + ch * 32, S_, lane), o0);
    o1 = wmma_bf16(ap, load_b_frag(vp + (size_t)16 * S_ + ch * 32, S_, lane), o1);
    o2 = wmma_bf16(ap, load_b_frag(vp + (size_t)32 * S_ + ch * 32, S_, lane), o2);
  }

  // ---- write head_cat slice (cols h*42 .. h*42+41) ----
  bf16* hr = hc + ((size_t)b * S_ + (size_t)itile * 16) * D_ + h * HS_;
#pragma unroll
  for (int r = 0; r < 8; ++r) {
    const int m = mb + r;
    hr[(size_t)m * D_ + n]       = (bf16)o0[r];
    hr[(size_t)m * D_ + 16 + n]  = (bf16)o1[r];
    if (32 + n < HS_) hr[(size_t)m * D_ + 32 + n] = (bf16)o2[r];
  }
}

// ---------------------------------------------------------------------------
// Dense WMMA GEMM with async-LDS staging.
// C[M,N] = A[M,K](bf16) @ Bt[N,K](bf16 transposed) + bias, optional ReLU.
// Block = 128 threads (4 waves) computes a 128x64 tile; wave w owns rows
// [w*32, w*32+32). Per 32-wide K step the 128x32 A panel and 64x32 B panel
// are DMA'd to LDS (global_load_async_to_lds_b128, double-buffered, ASYNCcnt:
// 6 issues/thread/stage), then each wave does 8 WMMAs off 2 A + 4 B LDS
// fragments. K-step count is a template param so the pipeline fully unrolls.
// grid = (M/128, N/64)
// ---------------------------------------------------------------------------
__device__ __forceinline__ void stage_panels(const bf16* Ab, int lda,
                                             const bf16* Bb, int ldb, int k0,
                                             bf16 (*sA)[32], bf16 (*sB)[32], int t) {
  // A: 128 rows x 32 bf16 = 8 KB = 512 x 16B transfers (4/thread)
#pragma unroll
  for (int j = 0; j < 4; ++j) {
    const int i = t + 128 * j;          // 0..511
    const int row = i >> 2, seg = (i & 3) * 8;
    async_copy_b128(lds_addr_of(&sA[row][seg]), Ab + (size_t)row * lda + k0 + seg);
  }
  // B: 64 rows x 32 bf16 = 4 KB = 256 x 16B transfers (2/thread)
#pragma unroll
  for (int j = 0; j < 2; ++j) {
    const int i = t + 128 * j;          // 0..255
    const int row = i >> 2, seg = (i & 3) * 8;
    async_copy_b128(lds_addr_of(&sB[row][seg]), Bb + (size_t)row * ldb + k0 + seg);
  }
}

template <bool RELU, int NK>
__global__ __launch_bounds__(128) void gemm_kernel(
    const bf16* __restrict__ A, int lda, const bf16* __restrict__ Bt, int ldb,
    const float* __restrict__ bias, float* Cf, bf16* Cb, int ldc) {
  __shared__ __align__(16) bf16 sA[2][128][32];  // 16 KB double-buffered A panel
  __shared__ __align__(16) bf16 sB[2][64][32];   //  8 KB double-buffered B panel

  const int t    = threadIdx.x;
  const int lane = t & 31;
  const int w    = t >> 5;                       // wave id: rows [w*32, w*32+32)
  const size_t row0 = (size_t)blockIdx.x * 128;
  const int    n0   = blockIdx.y * 64;
  const bf16* Ab = A + row0 * lda;
  const bf16* Bb = Bt + (size_t)n0 * ldb;

  v8f acc0[4] = {};   // rows w*32 .. w*32+15
  v8f acc1[4] = {};   // rows w*32+16 .. w*32+31

  stage_panels(Ab, lda, Bb, ldb, 0, sA[0], sB[0], t);
#pragma unroll
  for (int kk = 0; kk < NK; ++kk) {
    const int cur = kk & 1;
    if (kk + 1 < NK) {
      // prior compute on buffer cur^1 finished at last iteration's barrier
      stage_panels(Ab, lda, Bb, ldb, (kk + 1) * 32, sA[cur ^ 1], sB[cur ^ 1], t);
      wait_asynccnt<6>();              // stage kk retired (async done in-order)
    } else {
      wait_asynccnt<0>();
    }
    __syncthreads();                   // all waves' DMA for buffer cur visible

    const v16bf a0 = load_a_frag(&sA[cur][w * 32][0], 32, lane);       // ds_load
    const v16bf a1 = load_a_frag(&sA[cur][w * 32 + 16][0], 32, lane);
#pragma unroll
    for (int nt = 0; nt < 4; ++nt) {
      const v16bf bfr = load_b_frag(&sB[cur][nt * 16][0], 32, lane);
      acc0[nt] = wmma_bf16(a0, bfr, acc0[nt]);
      acc1[nt] = wmma_bf16(a1, bfr, acc1[nt]);
    }
    __syncthreads();                   // done reading buffer cur before reuse
  }

  const int n  = lane & 15;
  const int mb = (lane & 16) ? 8 : 0;
#pragma unroll
  for (int nt = 0; nt < 4; ++nt) {
    const float bn = bias ? bias[n0 + nt * 16 + n] : 0.f;
#pragma unroll
    for (int r = 0; r < 8; ++r) {
      float v0 = acc0[nt][r] + bn;
      float v1 = acc1[nt][r] + bn;
      if (RELU) { v0 = fmaxf(v0, 0.f); v1 = fmaxf(v1, 0.f); }
      const size_t i0 = (row0 + w * 32 + mb + r) * (size_t)ldc + n0 + nt * 16 + n;
      const size_t i1 = i0 + (size_t)16 * ldc;
      if (Cf) { Cf[i0] = v0; Cf[i1] = v1; }
      if (Cb) { Cb[i0] = (bf16)v0; Cb[i1] = (bf16)v1; }
    }
  }
}

// ---------------------------------------------------------------------------
// Residual add + LayerNorm over D=256. One 256-thread block (8 waves) per row.
// ---------------------------------------------------------------------------
__global__ __launch_bounds__(256) void resid_ln_kernel(
    const float* __restrict__ a, const float* __restrict__ c,
    const float* __restrict__ gamma, const float* __restrict__ beta,
    float* __restrict__ outf, bf16* outb) {
  __shared__ float red[8];
  const int row = blockIdx.x, t = threadIdx.x;
  const size_t off = (size_t)row * D_ + t;
  const float v = a[off] + c[off];

  float s = v;
#pragma unroll
  for (int o = 16; o; o >>= 1) s += __shfl_xor(s, o, 32);
  if ((t & 31) == 0) red[t >> 5] = s;
  __syncthreads();
  float mean = 0.f;
#pragma unroll
  for (int i = 0; i < 8; ++i) mean += red[i];
  mean *= (1.f / D_);
  __syncthreads();

  const float d = v - mean;
  float s2 = d * d;
#pragma unroll
  for (int o = 16; o; o >>= 1) s2 += __shfl_xor(s2, o, 32);
  if ((t & 31) == 0) red[t >> 5] = s2;
  __syncthreads();
  float var = 0.f;
#pragma unroll
  for (int i = 0; i < 8; ++i) var += red[i];
  var *= (1.f / D_);

  const float r = rsqrtf(var + 1e-5f);
  const float o = d * r * gamma[t] + beta[t];
  outf[off] = o;
  if (outb) outb[off] = (bf16)o;
}

// ---------------------------------------------------------------------------
// Host launch
// ---------------------------------------------------------------------------
extern "C" void kernel_launch(void* const* d_in, const int* in_sizes, int n_in,
                              void* d_out, int out_size, void* d_ws, size_t ws_size,
                              hipStream_t stream) {
  (void)in_sizes; (void)n_in; (void)out_size; (void)ws_size;
  const float* x     = (const float*)d_in[0];
  const float* Wq    = (const float*)d_in[1];
  const float* Wk    = (const float*)d_in[2];
  const float* Wv    = (const float*)d_in[3];
  const float* Wproj = (const float*)d_in[4];
  const float* bproj = (const float*)d_in[5];
  const float* ln1g  = (const float*)d_in[6];
  const float* ln1b  = (const float*)d_in[7];
  const float* W1    = (const float*)d_in[8];
  const float* b1    = (const float*)d_in[9];
  const float* W2    = (const float*)d_in[10];
  const float* b2    = (const float*)d_in[11];
  const float* ln2g  = (const float*)d_in[12];
  const float* ln2b  = (const float*)d_in[13];
  float* out = (float*)d_out;

  char* w = (char*)d_ws;
  auto alloc = [&](size_t bytes) -> void* {
    void* p = (void*)w;
    w += (bytes + 255) & ~(size_t)255;
    return p;
  };
  bf16*  xb   = (bf16*) alloc((size_t)BS_ * D_ * 2);          // x in bf16
  bf16*  qb   = (bf16*) alloc((size_t)B_ * H_ * S_ * HSP_ * 2);
  bf16*  kb   = (bf16*) alloc((size_t)B_ * H_ * S_ * HSP_ * 2);
  bf16*  vTb  = (bf16*) alloc((size_t)B_ * H_ * HSP_ * S_ * 2);
  bf16*  hc   = (bf16*) alloc((size_t)BS_ * D_ * 2);          // head_cat (252 pad 256)
  float* tmp  = (float*)alloc((size_t)BS_ * D_ * 4);          // matmul f32 out (reused)
  float* l1f  = (float*)alloc((size_t)BS_ * D_ * 4);          // ln1 out f32
  bf16*  l1bf = (bf16*) alloc((size_t)BS_ * D_ * 2);          // ln1 out bf16
  bf16*  ffb  = (bf16*) alloc((size_t)BS_ * DFF_ * 2);        // relu(ffn1) bf16
  bf16*  WqT  = (bf16*) alloc((size_t)H_ * HSP_ * D_ * 2);
  bf16*  WkT  = (bf16*) alloc((size_t)H_ * HSP_ * D_ * 2);
  bf16*  WvT  = (bf16*) alloc((size_t)H_ * HSP_ * D_ * 2);
  bf16*  WpT  = (bf16*) alloc((size_t)D_ * D_ * 2);
  bf16*  W1T  = (bf16*) alloc((size_t)DFF_ * D_ * 2);
  bf16*  W2T  = (bf16*) alloc((size_t)D_ * DFF_ * 2);

  // --- prep / pack ---
  pack_x_kernel<<<(BS_ * D_ + 255) / 256, 256, 0, stream>>>(x, xb, BS_ * D_);
  pack_qkvw_kernel<<<(H_ * HSP_ * D_ + 255) / 256, 256, 0, stream>>>(Wq, Wk, Wv, WqT, WkT, WvT);
  pack_proj_kernel<<<(D_ * D_ + 255) / 256, 256, 0, stream>>>(Wproj, WpT);
  pack_t_kernel<<<(D_ * DFF_ + 255) / 256, 256, 0, stream>>>(W1, W1T, D_, DFF_);
  pack_t_kernel<<<(DFF_ * D_ + 255) / 256, 256, 0, stream>>>(W2, W2T, DFF_, D_);
  zero_hc_tail_kernel<<<(BS_ * 4 + 255) / 256, 256, 0, stream>>>(hc);

  // --- attention path ---
  qkv_kernel<<<dim3(S_ / 16, HSP_ / 16, B_ * H_), 32, 0, stream>>>(xb, WqT, WkT, WvT, qb, kb, vTb);
  attn_kernel<<<dim3(S_ / 16, B_ * H_), 32, 0, stream>>>(qb, kb, vTb, hc);
  gemm_kernel<false, D_ / 32><<<dim3(BS_ / 128, D_ / 64), 128, 0, stream>>>(
      hc, D_, WpT, D_, bproj, tmp, nullptr, D_);
  resid_ln_kernel<<<BS_, 256, 0, stream>>>(x, tmp, ln1g, ln1b, l1f, l1bf);

  // --- FFN path ---
  gemm_kernel<true, D_ / 32><<<dim3(BS_ / 128, DFF_ / 64), 128, 0, stream>>>(
      l1bf, D_, W1T, D_, b1, nullptr, ffb, DFF_);
  gemm_kernel<false, DFF_ / 32><<<dim3(BS_ / 128, D_ / 64), 128, 0, stream>>>(
      ffb, DFF_, W2T, DFF_, b2, tmp, nullptr, D_);
  resid_ln_kernel<<<BS_, 256, 0, stream>>>(l1f, tmp, ln2g, ln2b, out, nullptr);
}
